// GINDecoder_43241730736197
// MI455X (gfx1250) — compile-verified, matrix operand order checked by reading, not run on hardware
//
#include <hip/hip_runtime.h>
#include <hip/hip_bf16.h>

// ---------------------------------------------------------------------------
// GIN (3 layers) + bilinear decoder for MI455X (gfx1250, wave32, WMMA).
//
// Cost model: edge scatter-add dominates (3 x ~410MB gather + 102M f32
// atomics, all L2-resident: node features are 25.6MB << 192MB L2). The MLP
// GEMMs (~20 GFLOP) run on v_wmma_f32_16x16x32_bf16 with f32 accumulation.
// Both operands are PRE-PACKED into the ISA 7.12.2 fragment layouts as bf16,
// so the GEMM inner loop is just two 32B contiguous loads + one WMMA per
// K-chunk (no per-tile f32->bf16 conversion, no redundant weight refetch).
// ---------------------------------------------------------------------------

#define GIN_N_NODES 50000
#define GIN_N_EDGES 800000
#define GIN_D_IN    128
#define GIN_D_HID2  256   // 2*D_IN hidden width of the GIN MLP
#define GIN_BATCH   4096
#define GIN_EPS     0.01f

typedef __attribute__((ext_vector_type(16))) __bf16 v16bf;
typedef __attribute__((ext_vector_type(8)))  float  v8f;

// f32 -> bf16 bit pattern, round-to-nearest-even.
__device__ __forceinline__ unsigned short f2bf_bits(float f) {
    union { float f; unsigned u; } x;
    x.f = f;
    unsigned r = x.u + 0x7FFFu + ((x.u >> 16) & 1u);
    return (unsigned short)(r >> 16);
}

// K offset of VGPR-pair v (0..7) for half-wave g, per ISA 7.12.2 16-bit A/B
// fragment layout: VGPR v holds K = kbase(v,g) + {0,1}.
__device__ __forceinline__ int kbase(int v, int g) {
    return ((v < 4) ? (2 * v) : (16 + 2 * (v - 4))) + 8 * g;
}

// ---------------------------------------------------------------------------
// Zero fill (aggregation buffer).
// ---------------------------------------------------------------------------
__global__ void gin_zero_f32(float* __restrict__ p, long n) {
    long i = (long)blockIdx.x * blockDim.x + threadIdx.x;
    const long stride = (long)gridDim.x * blockDim.x;
    for (; i < n; i += stride) p[i] = 0.0f;
}

// ---------------------------------------------------------------------------
// Edge scatter-add: one wave per edge, 32 lanes x float4 = 128 features.
// agg[dst] += h[src]. All lanes share (src,dst) -> scalar broadcast loads.
// ---------------------------------------------------------------------------
__global__ void gin_edge_scatter(const float* __restrict__ H,
                                 const int* __restrict__ src,
                                 const int* __restrict__ dst,
                                 float* __restrict__ AGG) {
    const int e    = (int)((blockIdx.x * blockDim.x + threadIdx.x) >> 5);
    const int lane = threadIdx.x & 31;
    if (e >= GIN_N_EDGES) return;
    const int s = src[e];
    const int d = dst[e];
    const float4 val = ((const float4*)(H + (size_t)s * GIN_D_IN))[lane];
    float* o = AGG + (size_t)d * GIN_D_IN + lane * 4;
    atomicAdd(o + 0, val.x);
    atomicAdd(o + 1, val.y);
    atomicAdd(o + 2, val.z);
    atomicAdd(o + 3, val.w);
}

// ---------------------------------------------------------------------------
// Pack an f32 activation matrix [M x K] into bf16 A-fragment layout.
// Fragment f = rowtile*(K/32) + kchunk; per fragment, lane L (0..31) owns 16
// contiguous bf16 (32B) at out[(f*32+L)*16]. Optional fused GIN combine:
// (1+eps)*X + AGG. One thread per (fragment, lane): reads 8 float2 pairs,
// converts once, writes one 32B chunk.
// ---------------------------------------------------------------------------
template <bool COMBINE>
__global__ void gin_pack_a(const float* __restrict__ X,
                           const float* __restrict__ AGG,
                           unsigned short* __restrict__ out,
                           int M, int K) {
    const int KC  = K >> 5;
    const int tid = blockIdx.x * blockDim.x + threadIdx.x;
    const int total = (M >> 4) * KC * 32;
    if (tid >= total) return;
    const int lane = tid & 31;
    const int f    = tid >> 5;
    const int kc   = f % KC;
    const int rt   = f / KC;
    const int g    = lane >> 4;
    const int row  = rt * 16 + (lane & 15);
    const float* xr = X + (size_t)row * K;
    const float* gr = COMBINE ? (AGG + (size_t)row * K) : nullptr;

    unsigned short vals[16];
#pragma unroll
    for (int v = 0; v < 8; ++v) {
        const int k = kc * 32 + kbase(v, g);
        float a0 = xr[k];
        float a1 = xr[k + 1];
        if constexpr (COMBINE) {
            a0 = (1.0f + GIN_EPS) * a0 + gr[k];
            a1 = (1.0f + GIN_EPS) * a1 + gr[k + 1];
        }
        vals[2 * v]     = f2bf_bits(a0);
        vals[2 * v + 1] = f2bf_bits(a1);
    }
    __builtin_memcpy(out + (size_t)tid * 16, vals, 32);
}

// ---------------------------------------------------------------------------
// Pack an f32 weight matrix [K x N] into bf16 B-fragment layout.
// Fragment f = ntile*(K/32) + kchunk; lane L owns column n = ntile*16 + L%16,
// elements cover K = kbase(e/2, L/16) + e%2. Runs once per launch (tiny).
// ---------------------------------------------------------------------------
__global__ void gin_pack_w(const float* __restrict__ W,
                           unsigned short* __restrict__ out,
                           int K, int N) {
    const int KC  = K >> 5;
    const int tid = blockIdx.x * blockDim.x + threadIdx.x;
    const int total = (N >> 4) * KC * 32;
    if (tid >= total) return;
    const int lane = tid & 31;
    const int f    = tid >> 5;
    const int kc   = f % KC;
    const int tn   = f / KC;
    const int g    = lane >> 4;
    const int col  = tn * 16 + (lane & 15);

    unsigned short vals[16];
#pragma unroll
    for (int v = 0; v < 8; ++v) {
        const int k = kc * 32 + kbase(v, g);
        vals[2 * v]     = f2bf_bits(W[(size_t)k * N + col]);
        vals[2 * v + 1] = f2bf_bits(W[(size_t)(k + 1) * N + col]);
    }
    __builtin_memcpy(out + (size_t)tid * 16, vals, 32);
}

// ---------------------------------------------------------------------------
// WMMA GEMM on pre-packed fragments: C[M x N] = act(A @ W + bias).
// One wave per 16x16 output tile; per K-chunk: one 32B A load + one 32B B
// load per lane + one v_wmma_f32_16x16x32_bf16. Grids divide exactly, so
// EXEC is all-1s at every WMMA (ISA requirement).
// ---------------------------------------------------------------------------
template <bool RELU>
__global__ void gin_gemm_wmma(const unsigned short* __restrict__ Afrag,
                              const unsigned short* __restrict__ Wfrag,
                              const float* __restrict__ bias,
                              float* __restrict__ C,
                              int M, int K, int N) {
    const int lane      = threadIdx.x & 31;
    const int waveInBlk = threadIdx.x >> 5;
    const int gtile     = blockIdx.x * (blockDim.x >> 5) + waveInBlk;
    const int tilesN    = N >> 4;
    const int nTiles    = (M >> 4) * tilesN;
    if (gtile >= nTiles) return;            // exact grids: never diverges

    const int tm = gtile / tilesN;
    const int tn = gtile % tilesN;
    const int KC = K >> 5;

    const v16bf* __restrict__ af = (const v16bf*)Afrag;
    const v16bf* __restrict__ bw = (const v16bf*)Wfrag;

    v8f acc = {};
    for (int kc = 0; kc < KC; ++kc) {
        const v16bf fa = af[((size_t)(tm * KC + kc)) * 32 + lane];
        const v16bf fb = bw[((size_t)(tn * KC + kc)) * 32 + lane];
        // (neg_a, A, neg_b, B, c_mod, C, reuse_a, reuse_b)
        acc = __builtin_amdgcn_wmma_f32_16x16x32_bf16(
            false, fa, false, fb, (short)0, acc, false, false);
    }

    const int g    = lane >> 4;
    const int lp   = lane & 15;
    const int colB = tn * 16 + lp;
    const float bcol = bias[colB];
#pragma unroll
    for (int r = 0; r < 8; ++r) {
        const int m = tm * 16 + r + 8 * g;  // C/D layout: VGPR r -> M = r + 8*half
        float o = acc[r] + bcol;
        if (RELU) o = fmaxf(o, 0.0f);
        C[(size_t)m * N + colB] = o;
    }
}

// ---------------------------------------------------------------------------
// Decoder: one 64-thread block per batch row.
//   a = h[i0], b = h[i1]; A = a@P1, Bv = b@P1 (128->64);
//   y = sum( (A@P2) * Bv )
// ---------------------------------------------------------------------------
__global__ void gin_decoder(const float* __restrict__ H,
                            const int* __restrict__ drug,
                            const float* __restrict__ P1,
                            const float* __restrict__ P2,
                            float* __restrict__ out) {
    __shared__ float sA[64];
    __shared__ float sB[64];
    __shared__ float sR[64];
    const int row = blockIdx.x;
    const int c   = threadIdx.x;            // 0..63 -> one D_DEC column each
    const int ia  = drug[row * 2 + 0] - 1;
    const int ib  = drug[row * 2 + 1] - 1;
    const float* a = H + (size_t)ia * GIN_D_IN;
    const float* b = H + (size_t)ib * GIN_D_IN;

    float accA = 0.0f, accB = 0.0f;
    for (int k = 0; k < GIN_D_IN; ++k) {
        const float p = P1[k * 64 + c];
        accA += a[k] * p;
        accB += b[k] * p;
    }
    sA[c] = accA;
    sB[c] = accB;
    __syncthreads();

    float t = 0.0f;
    for (int j = 0; j < 64; ++j) t += sA[j] * P2[j * 64 + c];
    sR[c] = t * sB[c];
    __syncthreads();

    for (int s = 32; s > 0; s >>= 1) {
        if (c < s) sR[c] += sR[c + s];
        __syncthreads();
    }
    if (c == 0) out[row] = sR[0];
}

// ---------------------------------------------------------------------------
// Host-side orchestration (graph-capture safe: only kernel launches on stream)
// ---------------------------------------------------------------------------
extern "C" void kernel_launch(void* const* d_in, const int* in_sizes, int n_in,
                              void* d_out, int out_size, void* d_ws, size_t ws_size,
                              hipStream_t stream) {
    (void)in_sizes; (void)n_in; (void)out_size; (void)ws_size;

    // setup_inputs order:
    // 0:x 1:edge_index 2:drug_index 3:label
    // 4:W1a 5:b1a 6:W2a 7:b2a  8:W1b 9:b1b 10:W2b 11:b2b
    // 12:W1c 13:b1c 14:W2c 15:b2c  16:P1 17:P2
    const float* x    = (const float*)d_in[0];
    const int*   eidx = (const int*)d_in[1];
    const int*   drug = (const int*)d_in[2];
    const float* W1[3] = {(const float*)d_in[4],  (const float*)d_in[8],  (const float*)d_in[12]};
    const float* b1[3] = {(const float*)d_in[5],  (const float*)d_in[9],  (const float*)d_in[13]};
    const float* W2[3] = {(const float*)d_in[6],  (const float*)d_in[10], (const float*)d_in[14]};
    const float* b2[3] = {(const float*)d_in[7],  (const float*)d_in[11], (const float*)d_in[15]};
    const float* P1   = (const float*)d_in[16];
    const float* P2   = (const float*)d_in[17];
    float* out = (float*)d_out;

    // Workspace layout (bytes, all 256-aligned):
    //   agg    @ 0           : 25,600,000   (N*128 f32)
    //   hhid   @ 25,600,000  : 51,200,000   (N*256 f32)
    //   h1     @ 76,800,000  : 25,600,000
    //   h2     @ 102,400,000 : 25,600,000
    //   afrag1 @ 128,000,000 : 12,800,000   (N*128 bf16, packed)
    //   afrag2 @ 140,800,000 : 25,600,000   (N*256 bf16, packed)
    //   wfrags @ 166,400,000 : 6 x 65,536   (packed bf16 weights)
    char* ws = (char*)d_ws;
    float*          agg    = (float*)(ws);
    float*          hhid   = (float*)(ws + 25600000u);
    float*          h1     = (float*)(ws + 76800000u);
    float*          h2     = (float*)(ws + 102400000u);
    unsigned short* afrag1 = (unsigned short*)(ws + 128000000u);
    unsigned short* afrag2 = (unsigned short*)(ws + 140800000u);
    unsigned short* w1f[3];
    unsigned short* w2f[3];
    for (int i = 0; i < 3; ++i) {
        w1f[i] = (unsigned short*)(ws + 166400000u + (size_t)i * 65536u);
        w2f[i] = (unsigned short*)(ws + 166400000u + 196608u + (size_t)i * 65536u);
    }

    const int* src = eidx;                  // edge_index[0, :]
    const int* dst = eidx + GIN_N_EDGES;    // edge_index[1, :]

    // ---- pack weights into bf16 B-fragment layout (2048 threads each) ----
    for (int i = 0; i < 3; ++i) {
        gin_pack_w<<<8, 256, 0, stream>>>(W1[i], w1f[i], GIN_D_IN,   GIN_D_HID2);
        gin_pack_w<<<8, 256, 0, stream>>>(W2[i], w2f[i], GIN_D_HID2, GIN_D_IN);
    }

    const long aggElems = (long)GIN_N_NODES * GIN_D_IN;              // 6.4M
    const int  edgeBlocks   = GIN_N_EDGES / 8;                       // 8 waves/block
    const int  packA1Blocks = ((GIN_N_NODES / 16) * (GIN_D_IN   / 32) * 32 + 255) / 256;
    const int  packA2Blocks = ((GIN_N_NODES / 16) * (GIN_D_HID2 / 32) * 32 + 255) / 256;
    const int  gemm1Blocks  = (GIN_N_NODES / 16) * (GIN_D_HID2 / 16) / 8;  // 6250
    const int  gemm2Blocks  = (GIN_N_NODES / 16) * (GIN_D_IN   / 16) / 8;  // 3125

    const float* cur = x;
    float* ping = h1;
    float* pong = h2;

    for (int layer = 0; layer < 3; ++layer) {
        gin_zero_f32<<<2048, 256, 0, stream>>>(agg, aggElems);
        gin_edge_scatter<<<edgeBlocks, 256, 0, stream>>>(cur, src, dst, agg);
        // afrag1 = pack_bf16( (1+eps)*cur + agg )            [50000 x 128]
        gin_pack_a<true><<<packA1Blocks, 256, 0, stream>>>(
            cur, agg, afrag1, GIN_N_NODES, GIN_D_IN);
        // hhid = relu( afrag1 @ W1 + b1 )                    [50000 x 256]
        gin_gemm_wmma<true><<<gemm1Blocks, 256, 0, stream>>>(
            afrag1, w1f[layer], b1[layer], hhid,
            GIN_N_NODES, GIN_D_IN, GIN_D_HID2);
        // afrag2 = pack_bf16( hhid )                         [50000 x 256]
        gin_pack_a<false><<<packA2Blocks, 256, 0, stream>>>(
            hhid, nullptr, afrag2, GIN_N_NODES, GIN_D_HID2);
        // ping = relu( afrag2 @ W2 + b2 )                    [50000 x 128]
        gin_gemm_wmma<true><<<gemm2Blocks, 256, 0, stream>>>(
            afrag2, w2f[layer], b2[layer], ping,
            GIN_N_NODES, GIN_D_HID2, GIN_D_IN);
        cur = ping;
        float* t = ping; ping = pong; pong = t;
    }

    gin_decoder<<<GIN_BATCH, 64, 0, stream>>>(cur, drug, P1, P2, out);
}
